// MultiAttention_44186623541601
// MI455X (gfx1250) — matrix operand order repaired and checked by reading, not compiled, not run on hardware
//
#include <hip/hip_runtime.h>
#include <hip/hip_bf16.h>

typedef __bf16 bf16;
typedef __attribute__((ext_vector_type(16))) __bf16 v16bf;
typedef __attribute__((ext_vector_type(8)))  float  v8f;
typedef __attribute__((ext_vector_type(4)))  int    v4i;

constexpr int CDIM = 512;
constexpr int MTOK = 27648;   // total tokens = B * D*W*H = 2*13824
constexpr int N3   = 1536;    // fused QKV width

#if defined(__AMDGCN__) && __has_builtin(__builtin_amdgcn_global_load_async_to_lds_b128)
#define USE_ASYNC_LDS 1
typedef __attribute__((address_space(1))) v4i* as1_v4i;
typedef __attribute__((address_space(3))) v4i* as3_v4i;
#else
#define USE_ASYNC_LDS 0
#endif

#if defined(__AMDGCN__) && __has_builtin(__builtin_amdgcn_s_wait_asynccnt)
#define WAIT_ASYNC() __builtin_amdgcn_s_wait_asynccnt(0)
#else
#define WAIT_ASYNC() asm volatile("s_wait_asynccnt 0x0" ::: "memory")
#endif

// ---------------------------------------------------------------- WMMA helpers
static __device__ inline v8f wmma_bf16(v16bf a, v16bf b, v8f c) {
  return __builtin_amdgcn_wmma_f32_16x16x32_bf16(false, a, false, b, (short)0, c, false, false);
}

// A fragment (16x32 bf16) from row-major LDS tile. ISA layout:
// lanes 0-15: m=lane, K = {0..7, 16..23}; lanes 16-31: m=lane-16, K = {8..15, 24..31}
static __device__ inline v16bf load_a(const bf16* s, int m0, int k0, int ld) {
  const int lane = threadIdx.x & 31;
  const bf16* r = s + (m0 + (lane & 15)) * ld + k0 + ((lane >> 4) << 3);
  union { v16bf v; uint4 u[2]; } t;
  t.u[0] = *(const uint4*)(r);
  t.u[1] = *(const uint4*)(r + 16);
  return t.v;
}

// B fragment (32x16 bf16) from row-major (K-major) LDS: n = lane%16, K = {8h..8h+7, 16+8h..16+8h+7}
static __device__ inline v16bf load_b_rm(const bf16* s, int k0, int n0, int ld) {
  const int lane = threadIdx.x & 31;
  const bf16* r = s + (k0 + ((lane >> 4) << 3)) * ld + n0 + (lane & 15);
  v16bf t;
#pragma unroll
  for (int i = 0; i < 8; i++) { t[i] = r[i * ld]; t[i + 8] = r[(i + 16) * ld]; }
  return t;
}

static __device__ inline int cm_of(int lane, int v) { return v + ((lane >> 4) << 3); }
static __device__ inline int cn_of(int lane)        { return lane & 15; }

// ---------------------------------------------------------------- small utility kernels
__global__ void cat3_bf16_k(const float* __restrict__ a, const float* __restrict__ b,
                            const float* __restrict__ c, bf16* __restrict__ dst,
                            int cols, long total) {
  long i = (long)blockIdx.x * 256 + threadIdx.x;
  if (i >= total) return;
  long row = i / (3 * cols);
  int  j   = (int)(i % (3 * cols));
  int sel = j / cols, jj = j % cols;
  const float* s = (sel == 0) ? a : (sel == 1 ? b : c);
  dst[i] = (bf16)s[row * cols + jj];
}

__global__ void cat3_f32_k(const float* __restrict__ a, const float* __restrict__ b,
                           const float* __restrict__ c, float* __restrict__ dst, int cols) {
  int i = blockIdx.x * 256 + threadIdx.x;
  if (i >= 3 * cols) return;
  int sel = i / cols, jj = i % cols;
  const float* s = (sel == 0) ? a : (sel == 1 ? b : c);
  dst[i] = s[jj];
}

__global__ void cvt_bf16_k(const float* __restrict__ s, bf16* __restrict__ d, long n) {
  long i = (long)blockIdx.x * 256 + threadIdx.x;
  if (i < n) d[i] = (bf16)s[i];
}

__global__ void zero_f32_k(float* __restrict__ d, int n) {
  int i = blockIdx.x * 256 + threadIdx.x;
  if (i < n) d[i] = 0.f;
}

// relative-position bias table: bias64[i*64+j] = pos_table[idx[j]-idx[i]+3]
__global__ void bias64_k(const float* __restrict__ pos, float* __restrict__ b64) {
  int i = blockIdx.x * 256 + threadIdx.x;
  if (i >= 4096) return;
  int qi = i >> 6, kj = i & 63;
  int a0 = qi >> 4, a1 = (qi >> 2) & 3, a2 = qi & 3;
  int b0 = kj >> 4, b1 = (kj >> 2) & 3, b2 = kj & 3;
  int r0 = b0 - a0 + 3, r1 = b1 - a1 + 3, r2 = b2 - a2 + 3;
  b64[i] = pos[(r0 * 7 + r1) * 7 + r2];
}

// ---------------------------------------------------------------- token-matrix builder
// X[(p*L + n)*512 + c] = x[p*512*L + c*L + n]  (batched transpose, LDS 32x32 tiles)
__global__ __launch_bounds__(256) void build_tokens_k(const float* __restrict__ x,
                                                      bf16* __restrict__ X, int L) {
  __shared__ float tile[32][33];
  const int p  = blockIdx.z;
  const int n0 = blockIdx.x * 32, c0 = blockIdx.y * 32;
  const int tx = threadIdx.x, ty = threadIdx.y;
  const long base = (long)p * CDIM * L;
#pragma unroll
  for (int i = 0; i < 4; i++) {
    int c = c0 + ty + i * 8, n = n0 + tx;
    if (n < L) tile[ty + i * 8][tx] = x[base + (long)c * L + n];
  }
  __syncthreads();
#pragma unroll
  for (int i = 0; i < 4; i++) {
    int n = n0 + ty + i * 8, c = c0 + tx;
    if (n < L) X[((long)p * L + n) * CDIM + c] = (bf16)tile[tx][ty + i * 8];
  }
}

// ---------------------------------------------------------------- WMMA GEMM
// C[M,N] = A[M,K](bf16) * W[K,N](bf16) + bias[N]
// Block tile 128x128, K-tile 64 (16 WMMAs per barrier); 8 waves, 64x32 per wave.
// MODE 0: store bf16; MODE 1: store f32; MODE 2: accumulate f32
template <int MODE>
__global__ __launch_bounds__(256) void gemm_k(const bf16* __restrict__ A,
                                              const bf16* __restrict__ W,
                                              const float* __restrict__ bias,
                                              void* __restrict__ outp,
                                              int M, int N, int K) {
  __shared__ bf16 As[128 * 64];
  __shared__ bf16 Bs[64 * 128];
  const int tid = threadIdx.x, wid = tid >> 5, lane = tid & 31;
  const int bm = blockIdx.y * 128, bn = blockIdx.x * 128;
  const int wm = (wid >> 2) * 64, wn = (wid & 3) * 32;

  v8f acc[4][2] = {};

  const int ar = tid >> 1, ac = (tid & 1) * 32;   // A: 128 rows x 64 cols, 32 elems/thread
  const int br = tid >> 2, bc = (tid & 3) * 32;   // B: 64 rows x 128 cols, 32 elems/thread
  const bf16* Aptr = A + (long)(bm + ar) * K + ac;
  const bf16* Wptr = W + (long)br * N + bn + bc;

  for (int k0 = 0; k0 < K; k0 += 64) {
#if USE_ASYNC_LDS
#pragma unroll
    for (int i = 0; i < 4; i++) {
      __builtin_amdgcn_global_load_async_to_lds_b128(
          (as1_v4i)(void*)(Aptr + k0 + i * 8),
          (as3_v4i)(void*)(As + ar * 64 + ac + i * 8), 0, 0);
      __builtin_amdgcn_global_load_async_to_lds_b128(
          (as1_v4i)(void*)(Wptr + (long)k0 * N + i * 8),
          (as3_v4i)(void*)(Bs + br * 128 + bc + i * 8), 0, 0);
    }
    WAIT_ASYNC();
#else
    {
      const uint4* ga = (const uint4*)(Aptr + k0);
      uint4* da = (uint4*)(As + ar * 64 + ac);
      da[0] = ga[0]; da[1] = ga[1]; da[2] = ga[2]; da[3] = ga[3];
      const uint4* gb = (const uint4*)(Wptr + (long)k0 * N);
      uint4* db = (uint4*)(Bs + br * 128 + bc);
      db[0] = gb[0]; db[1] = gb[1]; db[2] = gb[2]; db[3] = gb[3];
    }
#endif
    if (k0 + 64 < K) {
      __builtin_prefetch(Aptr + k0 + 64, 0, 0);
      __builtin_prefetch(Wptr + (long)(k0 + 64) * N, 0, 0);
    }
    __syncthreads();
#pragma unroll
    for (int ks = 0; ks < 2; ks++) {
      v16bf af[4], bfv[2];
#pragma unroll
      for (int mt = 0; mt < 4; mt++) af[mt] = load_a(As, wm + mt * 16, ks * 32, 64);
#pragma unroll
      for (int nt = 0; nt < 2; nt++) bfv[nt] = load_b_rm(Bs, ks * 32, wn + nt * 16, 128);
#pragma unroll
      for (int mt = 0; mt < 4; mt++)
#pragma unroll
        for (int nt = 0; nt < 2; nt++)
          acc[mt][nt] = wmma_bf16(af[mt], bfv[nt], acc[mt][nt]);
    }
    __syncthreads();
  }

#pragma unroll
  for (int mt = 0; mt < 4; mt++)
#pragma unroll
    for (int nt = 0; nt < 2; nt++)
#pragma unroll
      for (int v = 0; v < 8; v++) {
        int gm = bm + wm + mt * 16 + cm_of(lane, v);
        int gn = bn + wn + nt * 16 + cn_of(lane);
        float val = acc[mt][nt][v] + bias[gn];
        if      constexpr (MODE == 0) ((bf16*)outp)[(long)gm * N + gn] = (bf16)val;
        else if constexpr (MODE == 1) ((float*)outp)[(long)gm * N + gn] = val;
        else                          ((float*)outp)[(long)gm * N + gn] += val;
      }
}

// ---------------------------------------------------------------- branch 1: flash attention, seq 576
// grid(9 q-tiles, 8 heads, 48 batches), 128 threads (4 waves, 16 q-rows each)
// K is stored TRANSPOSED in LDS (Kt[d][j]) so QK^T B-fragments come from load_b_rm (per-element strided loads).
__global__ __launch_bounds__(128) void attn1_k(const bf16* __restrict__ QKV,
                                               bf16* __restrict__ AO) {
  __shared__ bf16 Qs[64 * 64], Kt[64 * 64], Vs[64 * 64];
  __shared__ float Ssh[4][16 * 64];
  __shared__ bf16  Ps[4][16 * 64];
  __shared__ float rowM[4][16], rowL[4][16], rowSc[4][16];

  const int p = blockIdx.z, h = blockIdx.y, qt = blockIdx.x;
  const int tid = threadIdx.x, wid = tid >> 5, lane = tid & 31;

  { // load Q tile (64 x 64)
    int r = tid >> 1, cb = (tid & 1) * 32;
    long R = (long)(p * 576 + qt * 64 + r);
    const uint4* src = (const uint4*)(QKV + R * N3 + h * 64 + cb);
    uint4* dst = (uint4*)(Qs + r * 64 + cb);
    dst[0] = src[0]; dst[1] = src[1]; dst[2] = src[2]; dst[3] = src[3];
  }
  if (lane < 16) { rowM[wid][lane] = -1e30f; rowL[wid][lane] = 0.f; }

  v8f oacc[4] = {};

  for (int jt = 0; jt < 9; jt++) {
    { // load K (transposed) and V tiles
      int r = tid >> 1, cb = (tid & 1) * 32;
      long R = (long)(p * 576 + jt * 64 + r);
      const uint4* sk = (const uint4*)(QKV + R * N3 + 512  + h * 64 + cb);
      const uint4* sv = (const uint4*)(QKV + R * N3 + 1024 + h * 64 + cb);
      union { uint4 u[4]; bf16 e[32]; } tk;
      tk.u[0] = sk[0]; tk.u[1] = sk[1]; tk.u[2] = sk[2]; tk.u[3] = sk[3];
      uint4* dv = (uint4*)(Vs + r * 64 + cb);
      dv[0] = sv[0]; dv[1] = sv[1]; dv[2] = sv[2]; dv[3] = sv[3];
#pragma unroll
      for (int i = 0; i < 32; i++) Kt[(cb + i) * 64 + r] = tk.e[i];
    }
    __syncthreads();
    // S = (Q K^T) * 1/sqrt(64)
#pragma unroll
    for (int nt = 0; nt < 4; nt++) {
      v8f s = {};
#pragma unroll
      for (int ks = 0; ks < 2; ks++) {
        v16bf a = load_a(Qs, wid * 16, ks * 32, 64);
        v16bf b = load_b_rm(Kt, ks * 32, nt * 16, 64);
        s = wmma_bf16(a, b, s);
      }
#pragma unroll
      for (int v = 0; v < 8; v++)
        Ssh[wid][cm_of(lane, v) * 64 + nt * 16 + cn_of(lane)] = s[v] * 0.125f;
    }
    __syncthreads();
    { // online softmax: 2 lanes per row
      int r = lane >> 1, half = lane & 1;
      const float* srow = &Ssh[wid][r * 64 + half * 32];
      float mx = -1e30f;
#pragma unroll
      for (int c = 0; c < 32; c++) mx = fmaxf(mx, srow[c]);
      mx = fmaxf(mx, __shfl_xor(mx, 1));
      float mold = rowM[wid][r];
      float mnew = fmaxf(mold, mx);
      float corr = __expf(mold - mnew);
      float ls = 0.f;
      bf16* prow = &Ps[wid][r * 64 + half * 32];
#pragma unroll
      for (int c = 0; c < 32; c++) { float pv = __expf(srow[c] - mnew); prow[c] = (bf16)pv; ls += pv; }
      ls += __shfl_xor(ls, 1);
      if (half == 0) {
        rowM[wid][r]  = mnew;
        rowL[wid][r]  = rowL[wid][r] * corr + ls;
        rowSc[wid][r] = corr;
      }
    }
    __syncthreads();
    // O = diag(corr)*O + P V
#pragma unroll
    for (int nt = 0; nt < 4; nt++) {
      v8f o = oacc[nt];
#pragma unroll
      for (int v = 0; v < 8; v++) o[v] *= rowSc[wid][cm_of(lane, v)];
#pragma unroll
      for (int ks = 0; ks < 2; ks++) {
        v16bf a = load_a(Ps[wid], 0, ks * 32, 64);
        v16bf b = load_b_rm(Vs, ks * 32, nt * 16, 64);
        o = wmma_bf16(a, b, o);
      }
      oacc[nt] = o;
    }
    __syncthreads();
  }
#pragma unroll
  for (int nt = 0; nt < 4; nt++)
#pragma unroll
    for (int v = 0; v < 8; v++) {
      int ml = cm_of(lane, v);
      long R = (long)(p * 576 + qt * 64 + wid * 16 + ml);
      float inv = 1.f / rowL[wid][ml];
      AO[R * CDIM + h * 64 + nt * 16 + cn_of(lane)] = (bf16)(oacc[nt][v] * inv);
    }
}

// ---------------------------------------------------------------- branch 2: seq 24 (padded to 32)
// 64 threads = 2 waves, one (batch, head) unit per wave; K stored transposed
__global__ __launch_bounds__(64) void attn2_k(const bf16* __restrict__ QKV,
                                              bf16* __restrict__ AO) {
  __shared__ bf16 Q2[2][32 * 64], K2t[2][64 * 32], V2[2][32 * 64];
  __shared__ float S2[2][32 * 32];
  __shared__ bf16  P2[2][32 * 32];
  __shared__ float rinv[2][32];

  const int tid = threadIdx.x, wid = tid >> 5, lane = tid & 31;
  const int unit = blockIdx.x * 2 + wid;
  const int p = unit >> 3, h = unit & 7;

  { // one row (64 bf16) per lane, zero-pad rows/cols 24..31
    uint4* dq = (uint4*)(Q2[wid] + lane * 64);
    uint4* dv = (uint4*)(V2[wid] + lane * 64);
    if (lane < 24) {
      long R = (long)(p * 24 + lane);
      const uint4* sq = (const uint4*)(QKV + R * N3 + h * 64);
      const uint4* sk = (const uint4*)(QKV + R * N3 + 512  + h * 64);
      const uint4* sv = (const uint4*)(QKV + R * N3 + 1024 + h * 64);
      union { uint4 u[8]; bf16 e[64]; } tk;
#pragma unroll
      for (int i = 0; i < 8; i++) { dq[i] = sq[i]; tk.u[i] = sk[i]; dv[i] = sv[i]; }
#pragma unroll
      for (int i = 0; i < 64; i++) K2t[wid][i * 32 + lane] = tk.e[i];
    } else {
      uint4 z = {0, 0, 0, 0};
#pragma unroll
      for (int i = 0; i < 8; i++) { dq[i] = z; dv[i] = z; }
      bf16 zb = (bf16)0.f;
#pragma unroll
      for (int i = 0; i < 64; i++) K2t[wid][i * 32 + lane] = zb;
    }
  }
  __syncthreads();
#pragma unroll
  for (int mt = 0; mt < 2; mt++)
#pragma unroll
    for (int nt = 0; nt < 2; nt++) {
      v8f s = {};
#pragma unroll
      for (int ks = 0; ks < 2; ks++) {
        v16bf a = load_a(Q2[wid], mt * 16, ks * 32, 64);
        v16bf b = load_b_rm(K2t[wid], ks * 32, nt * 16, 32);
        s = wmma_bf16(a, b, s);
      }
#pragma unroll
      for (int v = 0; v < 8; v++) {
        int r2 = mt * 16 + cm_of(lane, v), c2 = nt * 16 + cn_of(lane);
        S2[wid][r2 * 32 + c2] = (c2 < 24) ? s[v] * 0.125f : -1e30f;
      }
    }
  __syncthreads();
  { // softmax: one row per lane
    float mx = -1e30f;
#pragma unroll
    for (int c = 0; c < 32; c++) mx = fmaxf(mx, S2[wid][lane * 32 + c]);
    float sum = 0.f;
#pragma unroll
    for (int c = 0; c < 32; c++) {
      float pv = __expf(S2[wid][lane * 32 + c] - mx);
      P2[wid][lane * 32 + c] = (bf16)pv;
      sum += pv;
    }
    rinv[wid][lane] = 1.f / sum;
  }
  __syncthreads();
#pragma unroll
  for (int mt = 0; mt < 2; mt++)
#pragma unroll
    for (int nt = 0; nt < 4; nt++) {
      v8f o = {};
      v16bf a = load_a(P2[wid], mt * 16, 0, 32);
      v16bf b = load_b_rm(V2[wid], 0, nt * 16, 64);
      o = wmma_bf16(a, b, o);
#pragma unroll
      for (int v = 0; v < 8; v++) {
        int r2 = mt * 16 + cm_of(lane, v);
        if (r2 < 24)
          AO[((long)(p * 24 + r2)) * CDIM + h * 64 + nt * 16 + cn_of(lane)] =
              (bf16)(o[v] * rinv[wid][r2]);
      }
    }
}

// ---------------------------------------------------------------- branch 3: 4x4x4 window attention
// grid(216 windows, 8 heads, 2 batches), 128 threads (4 waves); K stored transposed
__global__ __launch_bounds__(128) void attn3_k(const bf16* __restrict__ QKV,
                                               bf16* __restrict__ AO,
                                               const float* __restrict__ bias64) {
  __shared__ bf16 Qs[64 * 64], Kt[64 * 64], Vs[64 * 64];
  __shared__ float Ssh[4][16 * 64];
  __shared__ bf16  Ps[4][16 * 64];
  __shared__ float rowI[4][16];
  __shared__ int rowIdx[64];

  const int bz = blockIdx.z, h = blockIdx.y, w = blockIdx.x;
  const int dw = w / 36, ww = (w / 6) % 6, hw = w % 6;
  const int tid = threadIdx.x, wid = tid >> 5, lane = tid & 31;

  { // gather 64 window tokens; K goes in transposed
    int t = tid >> 1, cb = (tid & 1) * 32;
    int a = t >> 4, b2 = (t >> 2) & 3, c2 = t & 3;
    long R = (long)bz * 13824 + ((long)(dw * 4 + a) * 24 + (ww * 4 + b2)) * 24 + (hw * 4 + c2);
    if ((tid & 1) == 0) rowIdx[t] = (int)R;
    const uint4* sq = (const uint4*)(QKV + R * N3 + h * 64 + cb);
    const uint4* sk = (const uint4*)(QKV + R * N3 + 512  + h * 64 + cb);
    const uint4* sv = (const uint4*)(QKV + R * N3 + 1024 + h * 64 + cb);
    uint4* dq = (uint4*)(Qs + t * 64 + cb);
    uint4* dv = (uint4*)(Vs + t * 64 + cb);
    union { uint4 u[4]; bf16 e[32]; } tk;
#pragma unroll
    for (int i = 0; i < 4; i++) { dq[i] = sq[i]; tk.u[i] = sk[i]; dv[i] = sv[i]; }
#pragma unroll
    for (int i = 0; i < 32; i++) Kt[(cb + i) * 64 + t] = tk.e[i];
  }
  __syncthreads();
#pragma unroll
  for (int nt = 0; nt < 4; nt++) {
    v8f s = {};
#pragma unroll
    for (int ks = 0; ks < 2; ks++) {
      v16bf a = load_a(Qs, wid * 16, ks * 32, 64);
      v16bf b = load_b_rm(Kt, ks * 32, nt * 16, 64);
      s = wmma_bf16(a, b, s);
    }
#pragma unroll
    for (int v = 0; v < 8; v++) {
      int ml = cm_of(lane, v), col = nt * 16 + cn_of(lane);
      Ssh[wid][ml * 64 + col] = s[v] * 0.125f + bias64[(wid * 16 + ml) * 64 + col];
    }
  }
  __syncthreads();
  { // softmax over 64 keys
    int r = lane >> 1, half = lane & 1;
    const float* srow = &Ssh[wid][r * 64 + half * 32];
    float mx = -1e30f;
#pragma unroll
    for (int c = 0; c < 32; c++) mx = fmaxf(mx, srow[c]);
    mx = fmaxf(mx, __shfl_xor(mx, 1));
    float sum = 0.f;
    bf16* prow = &Ps[wid][r * 64 + half * 32];
#pragma unroll
    for (int c = 0; c < 32; c++) { float pv = __expf(srow[c] - mx); prow[c] = (bf16)pv; sum += pv; }
    sum += __shfl_xor(sum, 1);
    if (half == 0) rowI[wid][r] = 1.f / sum;
  }
  __syncthreads();
#pragma unroll
  for (int nt = 0; nt < 4; nt++) {
    v8f o = {};
#pragma unroll
    for (int ks = 0; ks < 2; ks++) {
      v16bf a = load_a(Ps[wid], 0, ks * 32, 64);
      v16bf b = load_b_rm(Vs, ks * 32, nt * 16, 64);
      o = wmma_bf16(a, b, o);
    }
#pragma unroll
    for (int v = 0; v < 8; v++) {
      int ml = cm_of(lane, v);
      long R = rowIdx[wid * 16 + ml];
      AO[R * CDIM + h * 64 + nt * 16 + cn_of(lane)] = (bf16)(o[v] * rowI[wid][ml]);
    }
  }
}

// ---------------------------------------------------------------- launch
extern "C" void kernel_launch(void* const* d_in, const int* in_sizes, int n_in,
                              void* d_out, int out_size, void* d_ws, size_t ws_size,
                              hipStream_t stream) {
  const float* x        = (const float*)d_in[0];
  const float* vq_w     = (const float*)d_in[1];
  const float* vq_b     = (const float*)d_in[2];
  const float* vk_w     = (const float*)d_in[3];
  const float* vk_b     = (const float*)d_in[4];
  const float* vv_w     = (const float*)d_in[5];
  const float* vv_b     = (const float*)d_in[6];
  const float* vo_w     = (const float*)d_in[7];
  const float* vo_b     = (const float*)d_in[8];
  const float* hq_w     = (const float*)d_in[9];
  const float* hq_b     = (const float*)d_in[10];
  const float* hk_w     = (const float*)d_in[11];
  const float* hk_b     = (const float*)d_in[12];
  const float* hv_w     = (const float*)d_in[13];
  const float* hv_b     = (const float*)d_in[14];
  const float* ho_w     = (const float*)d_in[15];
  const float* ho_b     = (const float*)d_in[16];
  const float* win_qkv_w= (const float*)d_in[17];
  const float* win_out_w= (const float*)d_in[18];
  const float* win_out_b= (const float*)d_in[19];
  const float* pos_table= (const float*)d_in[20];

  char* ws = (char*)d_ws;
  size_t off = 0;
  auto carve = [&](size_t bytes) -> void* {
    void* p = ws + off;
    off = (off + bytes + 255) & ~(size_t)255;
    return p;
  };
  bf16*  W1   = (bf16*) carve((size_t)CDIM * N3 * 2);
  bf16*  W2   = (bf16*) carve((size_t)CDIM * N3 * 2);
  bf16*  W3   = (bf16*) carve((size_t)CDIM * N3 * 2);
  bf16*  Wo1  = (bf16*) carve((size_t)CDIM * CDIM * 2);
  bf16*  Wo2  = (bf16*) carve((size_t)CDIM * CDIM * 2);
  bf16*  Wo3  = (bf16*) carve((size_t)CDIM * CDIM * 2);
  float* Bq1  = (float*)carve((size_t)N3 * 4);
  float* Bq2  = (float*)carve((size_t)N3 * 4);
  float* Bz   = (float*)carve((size_t)N3 * 4);
  float* b64  = (float*)carve((size_t)4096 * 4);
  bf16*  X    = (bf16*) carve((size_t)MTOK * CDIM * 2);
  bf16*  QKV  = (bf16*) carve((size_t)MTOK * N3 * 2);
  bf16*  AO   = (bf16*) carve((size_t)MTOK * CDIM * 2);
  (void)off; (void)ws_size; (void)in_sizes; (void)n_in; (void)out_size;

  float* out = (float*)d_out;

  // ---- weight / bias preparation
  long wtot = (long)CDIM * N3;
  cat3_bf16_k<<<(wtot + 255) / 256, 256, 0, stream>>>(vq_w, vk_w, vv_w, W1, CDIM, wtot);
  cat3_bf16_k<<<(wtot + 255) / 256, 256, 0, stream>>>(hq_w, hk_w, hv_w, W2, CDIM, wtot);
  cvt_bf16_k<<<(wtot + 255) / 256, 256, 0, stream>>>(win_qkv_w, W3, wtot);
  long ptot = (long)CDIM * CDIM;
  cvt_bf16_k<<<(ptot + 255) / 256, 256, 0, stream>>>(vo_w, Wo1, ptot);
  cvt_bf16_k<<<(ptot + 255) / 256, 256, 0, stream>>>(ho_w, Wo2, ptot);
  cvt_bf16_k<<<(ptot + 255) / 256, 256, 0, stream>>>(win_out_w, Wo3, ptot);
  cat3_f32_k<<<6, 256, 0, stream>>>(vq_b, vk_b, vv_b, Bq1, CDIM);
  cat3_f32_k<<<6, 256, 0, stream>>>(hq_b, hk_b, hv_b, Bq2, CDIM);
  zero_f32_k<<<6, 256, 0, stream>>>(Bz, N3);
  bias64_k<<<16, 256, 0, stream>>>(pos_table, b64);

  dim3 blkT(32, 8);
  dim3 gQKV(N3 / 128, MTOK / 128);
  dim3 gOUT(CDIM / 128, MTOK / 128);

  // ---- branch 1: full attention over (48, 576, 512)
  build_tokens_k<<<dim3(18, 16, 48), blkT, 0, stream>>>(x, X, 576);
  gemm_k<0><<<gQKV, 256, 0, stream>>>(X, W1, Bq1, QKV, MTOK, N3, CDIM);
  attn1_k<<<dim3(9, 8, 48), 128, 0, stream>>>(QKV, AO);
  gemm_k<1><<<gOUT, 256, 0, stream>>>(AO, Wo1, vo_b, out, MTOK, CDIM, CDIM);

  // ---- branch 2: full attention over (1152, 24, 512)
  build_tokens_k<<<dim3(1, 16, 1152), blkT, 0, stream>>>(x, X, 24);
  gemm_k<0><<<gQKV, 256, 0, stream>>>(X, W2, Bq2, QKV, MTOK, N3, CDIM);
  attn2_k<<<4608, 64, 0, stream>>>(QKV, AO);
  gemm_k<2><<<gOUT, 256, 0, stream>>>(AO, Wo2, ho_b, out, MTOK, CDIM, CDIM);

  // ---- branch 3: 4x4x4 window attention with relative-position bias
  build_tokens_k<<<dim3(432, 16, 2), blkT, 0, stream>>>(x, X, 13824);
  gemm_k<0><<<gQKV, 256, 0, stream>>>(X, W3, Bz, QKV, MTOK, N3, CDIM);
  attn3_k<<<dim3(216, 8, 2), 128, 0, stream>>>(QKV, AO, b64);
  gemm_k<2><<<gOUT, 256, 0, stream>>>(AO, Wo3, win_out_b, out, MTOK, CDIM, CDIM);
}
